// Isolator_89369679495275
// MI455X (gfx1250) — matrix operand order repaired and checked by reading, not compiled, not run on hardware
//
#include <hip/hip_runtime.h>
#include <hip/hip_bf16.h>
#include <math.h>
#include <stdint.h>

typedef __attribute__((ext_vector_type(16))) _Float16 v16h;
typedef __attribute__((ext_vector_type(8)))  _Float16 v8h;
typedef __attribute__((ext_vector_type(8)))  float    v8f;

#define BN_EPS 1e-5f
#define PI_F 3.14159265358979323846f

// ---------------------------------------------------------------------------
// conv1 (k=5, stride s, dilation d, pad 2d) + BN + ReLU
// ---------------------------------------------------------------------------
__global__ __launch_bounds__(256) void conv5_bn_relu(
    const float* __restrict__ in, float* __restrict__ out,
    const float* __restrict__ w, const float* __restrict__ bias,
    const float* __restrict__ gamma, const float* __restrict__ beta,
    const float* __restrict__ mean, const float* __restrict__ var,
    int B, int Cin, int Tin, int Cout, int Tout, int stride, int dil)
{
    int idx = blockIdx.x * blockDim.x + threadIdx.x;
    int total = B * Cout * Tout;
    if (idx >= total) return;
    int t  = idx % Tout;
    int co = (idx / Tout) % Cout;
    int b  = idx / (Tout * Cout);

    float acc = bias[co];
    int base = t * stride - 2 * dil;
    for (int ci = 0; ci < Cin; ++ci) {
        const float* ip = in + (size_t)(b * Cin + ci) * Tin;
        const float* wp = w + (co * Cin + ci) * 5;
#pragma unroll
        for (int k = 0; k < 5; ++k) {
            int p = base + k * dil;
            if (p >= 0 && p < Tin) acc += ip[p] * wp[k];
        }
    }
    float inv = gamma[co] * rsqrtf(var[co] + BN_EPS);
    float y = (acc - mean[co]) * inv + beta[co];
    out[idx] = fmaxf(y, 0.0f);
}

// ---------------------------------------------------------------------------
// conv2 (k=5, stride 1, dilation d, pad 2d) + BN + residual + ReLU
// ---------------------------------------------------------------------------
__global__ __launch_bounds__(256) void conv5_bn_add_relu(
    const float* __restrict__ a, const float* __restrict__ h,
    float* __restrict__ out,
    const float* __restrict__ w2, const float* __restrict__ b2,
    const float* __restrict__ g2, const float* __restrict__ be2,
    const float* __restrict__ m2, const float* __restrict__ v2,
    const float* __restrict__ wds, const float* __restrict__ gds,
    const float* __restrict__ beds, const float* __restrict__ mds,
    const float* __restrict__ vds,
    int B, int Cout, int Tout, int dil,
    int CinH, int TinH, int strideH, int has_ds)
{
    int idx = blockIdx.x * blockDim.x + threadIdx.x;
    int total = B * Cout * Tout;
    if (idx >= total) return;
    int t  = idx % Tout;
    int co = (idx / Tout) % Cout;
    int b  = idx / (Tout * Cout);

    float acc = b2[co];
    int base = t - 2 * dil;
    for (int ci = 0; ci < Cout; ++ci) {
        const float* ap = a + (size_t)(b * Cout + ci) * Tout;
        const float* wp = w2 + (co * Cout + ci) * 5;
#pragma unroll
        for (int k = 0; k < 5; ++k) {
            int p = base + k * dil;
            if (p >= 0 && p < Tout) acc += ap[p] * wp[k];
        }
    }
    float inv2 = g2[co] * rsqrtf(v2[co] + BN_EPS);
    float y = (acc - m2[co]) * inv2 + be2[co];

    float r;
    if (has_ds) {
        float racc = 0.0f;
        int hp = t * strideH;
        for (int ci = 0; ci < CinH; ++ci)
            racc += h[(size_t)(b * CinH + ci) * TinH + hp] * wds[co * CinH + ci];
        float invd = gds[co] * rsqrtf(vds[co] + BN_EPS);
        r = (racc - mds[co]) * invd + beds[co];
    } else {
        r = h[(size_t)(b * Cout + co) * Tout + t];
    }
    out[idx] = fmaxf(y + r, 0.0f);
}

// ---------------------------------------------------------------------------
// 1x1 regressor (32 -> 6) + sigmoid
// ---------------------------------------------------------------------------
__global__ __launch_bounds__(256) void regressor_sigmoid(
    const float* __restrict__ h, const float* __restrict__ w,
    const float* __restrict__ bias, float* __restrict__ out,
    int B, int Cin, int F, int Cout)
{
    int idx = blockIdx.x * blockDim.x + threadIdx.x;
    int total = B * Cout * F;
    if (idx >= total) return;
    int f  = idx % F;
    int co = (idx / F) % Cout;
    int b  = idx / (F * Cout);
    float acc = bias[co];
    for (int ci = 0; ci < Cin; ++ci)
        acc += h[(size_t)(b * Cin + ci) * F + f] * w[co * Cin + ci];
    out[idx] = 1.0f / (1.0f + __expf(-acc));
}

// ---------------------------------------------------------------------------
// FIR synthesis + sliding-window einsum via WMMA + volume interpolation.
// One wave32 per (b, f). Interior waves stage their x-span with async
// global->LDS copies (ASYNCcnt). The span is then converted once to f16 in 8
// shifted replicas so every A fragment is an aligned ds_load_b128; filter taps
// are synthesized straight to f16 so B fragments are two aligned v8h loads.
// ---------------------------------------------------------------------------
__global__ __launch_bounds__(256) void fir_wmma(
    const float* __restrict__ x, const float* __restrict__ h6,
    float* __restrict__ out, int B, int T, int F)
{
    __shared__ __align__(32) float    sspan32[8][320];     // raw f32 span
    __shared__ __align__(32) _Float16 srep[8][8][288];     // rep[s][j] = span[j+s]
    __shared__ __align__(32) _Float16 sfilth[8][3][256];   // [n0, n1, zeros]

    const int lane = threadIdx.x & 31;
    const int wv   = threadIdx.x >> 5;
    const int gw   = blockIdx.x * 8 + wv;
    const int b    = gw / F;
    const int f    = gw % F;

    // lo/hi per target from the faithful [B,6,F] -> [B,F,2,3] flat reshape
    const size_t hbase = (size_t)b * 6 * F + 6 * (size_t)f;
    float l0 = 0.5f * h6[hbase + 0];
    float u0 = l0 + h6[hbase + 1] * (0.5f - l0);
    float l1 = 0.5f * h6[hbase + 3];
    float u1 = l1 + h6[hbase + 4] * (0.5f - l1);

    // synthesize Hamming-windowed band-pass taps directly in f16
#pragma unroll
    for (int j = 0; j < 16; ++j) {
        int kidx = lane + 32 * j;        // 0..511
        int n = kidx >> 8;
        int k = kidx & 255;
        float val = 0.0f;
        if (k < 255) {
            float lo = n ? l1 : l0;
            float hi = n ? u1 : u0;
            float fac = -PI_F + (2.0f * PI_F / 254.0f) * (float)k;
            float win = 0.54f + 0.46f * __cosf(fac);
            float m   = (float)k - 127.0f;
            float lo2 = 2.0f * lo, hi2 = 2.0f * hi;
            float thi = hi2 * m,  tlo = lo2 * m;
            float shi = (thi == 0.0f) ? 1.0f : __sinf(PI_F * thi) / (PI_F * thi);
            float slo = (tlo == 0.0f) ? 1.0f : __sinf(PI_F * tlo) / (PI_F * tlo);
            val = (hi2 * shi - lo2 * slo) * win;
        }
        sfilth[wv][n][k] = (_Float16)val;
    }
#pragma unroll
    for (int j = 0; j < 8; ++j) sfilth[wv][2][lane + 32 * j] = (_Float16)0.0f;

    // stage x span: span[i] = x[f*32 - 127 + i] (zero outside [0,T))
    const int g0 = f * 32 - 127;
    if (g0 >= 0 && (g0 + 320) <= T) {
        // interior fast path: async global -> LDS copies, tracked by ASYNCcnt
        const float* gp = x + (size_t)b * T + g0;
#pragma unroll
        for (int rr = 0; rr < 10; ++rr) {
            int i = rr * 32 + lane;
            uint32_t loff  = (uint32_t)(uintptr_t)&sspan32[wv][i];
            uint64_t gaddr = (uint64_t)(uintptr_t)(gp + i);
            asm volatile("global_load_async_to_lds_b32 %0, %1, off"
                         :: "v"(loff), "v"(gaddr) : "memory");
        }
        asm volatile("s_wait_asynccnt 0x0" ::: "memory");
    } else {
        for (int i = lane; i < 320; i += 32) {
            int g = g0 + i;
            sspan32[wv][i] = (g >= 0 && g < T) ? x[(size_t)b * T + g] : 0.0f;
        }
    }

    // build 8 shifted f16 replicas so A fragments are 16B-aligned b128 loads
#pragma unroll
    for (int s = 0; s < 8; ++s)
        for (int j = lane; j < 288; j += 32)
            srep[wv][s][j] = (_Float16)sspan32[wv][j + s];

    __syncthreads();

    const int halfsel = lane >> 4;       // 0: lanes 0-15, 1: lanes 16-31
    const int mrow    = lane & 15;
    const int brow    = (mrow < 2) ? mrow : 2;
    const int srow    = mrow & 7;        // replica shift (same for both tiles)

    v8f acc0 = {0.f, 0.f, 0.f, 0.f, 0.f, 0.f, 0.f, 0.f};
    v8f acc1 = {0.f, 0.f, 0.f, 0.f, 0.f, 0.f, 0.f, 0.f};

    for (int kk = 0; kk < 8; ++kk) {
        // B 32x16 f16: lane L holds N=L%16, K = kk*32 + halfsel*8 + {0..7} and +16..23
        v8h b0 = *(const v8h*)&sfilth[wv][brow][kk * 32 + halfsel * 8];
        v8h b1 = *(const v8h*)&sfilth[wv][brow][kk * 32 + halfsel * 8 + 16];
        v16h bm = __builtin_shufflevector(b0, b1, 0,1,2,3,4,5,6,7,8,9,10,11,12,13,14,15);

        // A 16x32 f16: lane L holds M, halves split into two 8-tap runs:
        //   K = kk*32 + halfsel*8 + {0..7} and +16..23 -> span idx = M + K.
        // tile0-hi run == tile1-lo run, so 3 aligned v8h loads cover both tiles.
        int j0 = mrow + kk * 32 + halfsel * 8 - srow;     // multiple of 8
        v8h r0 = *(const v8h*)&srep[wv][srow][j0];        // tile0 lo
        v8h r1 = *(const v8h*)&srep[wv][srow][j0 + 16];   // tile0 hi / tile1 lo
        v8h r2 = *(const v8h*)&srep[wv][srow][j0 + 32];   // tile1 hi
        v16h a0 = __builtin_shufflevector(r0, r1, 0,1,2,3,4,5,6,7,8,9,10,11,12,13,14,15);
        v16h a1 = __builtin_shufflevector(r1, r2, 0,1,2,3,4,5,6,7,8,9,10,11,12,13,14,15);

        acc0 = __builtin_amdgcn_wmma_f32_16x16x32_f16(
            false, a0, false, bm, (short)0, acc0, false, false);
        acc1 = __builtin_amdgcn_wmma_f32_16x16x32_f16(
            false, a1, false, bm, (short)0, acc1, false, false);
    }

    // Epilogue: D layout lane L: N=L%16, VGPR r -> M = r + 8*(L/16) (+16 tile 1).
    const int n = mrow;
    if (n < 2) {
        const size_t vbase = (size_t)b * 6 * F;
#pragma unroll
        for (int tile = 0; tile < 2; ++tile) {
#pragma unroll
            for (int r = 0; r < 8; ++r) {
                int M = tile * 16 + halfsel * 8 + r;
                int t = f * 32 + M;
                float val = tile ? acc1[r] : acc0[r];
                float src = ((float)t + 0.5f) * (1.0f / 32.0f) - 0.5f;
                src = fminf(fmaxf(src, 0.0f), (float)(F - 1));
                int   i0 = (int)src;
                float fr = src - (float)i0;
                int   i1 = (i0 + 1 < F) ? (i0 + 1) : (F - 1);
                float v0 = h6[vbase + 6 * (size_t)i0 + 3 * n + 2];
                float v1 = h6[vbase + 6 * (size_t)i1 + 3 * n + 2];
                float vol = v0 + (v1 - v0) * fr;
                out[((size_t)(b * 2 + n)) * T + t] = val * vol;
            }
        }
    }
}

// ---------------------------------------------------------------------------
// Host launcher
// ---------------------------------------------------------------------------
extern "C" void kernel_launch(void* const* d_in, const int* in_sizes, int n_in,
                              void* d_out, int out_size, void* d_ws, size_t ws_size,
                              hipStream_t stream)
{
    (void)in_sizes; (void)n_in; (void)out_size; (void)ws_size;
    const int B = 4, T = 131072, F = 4096;

    const float* x = (const float*)d_in[0];
    float* ws   = (float*)d_ws;
    float* bufA = ws;                 // conv1 intermediate (<= 524288 floats)
    float* buf0 = ws + 524288;        // layer output ping
    float* buf1 = ws + 1048576;       // layer output pong
    float* h6   = ws + 1572864;       // regressor output [B,6,F]

    static const int cinL[7]  = {1, 2, 4, 4, 8, 8, 16};
    static const int coutL[7] = {2, 4, 4, 8, 8, 16, 32};
    static const int sL[7]    = {2, 2, 1, 2, 2, 1, 2};
    static const int dL[7]    = {1, 2, 2, 1, 2, 2, 1};
    static const int dsL[7]   = {1, 1, 0, 1, 1, 1, 1};
    static const int baseL[7] = {1, 18, 35, 47, 64, 81, 98};

    const float* cur = x;
    float* nxt = buf0;
    int Tin = T;

    for (int L = 0; L < 7; ++L) {
        int cin = cinL[L], cout = coutL[L], s = sL[L], d = dL[L];
        int Tout = (s == 2) ? (Tin / 2) : Tin;
        int base = baseL[L];
        auto P = [&](int k) { return (const float*)d_in[base + k]; };

        int tot1 = B * cout * Tout;
        conv5_bn_relu<<<(tot1 + 255) / 256, 256, 0, stream>>>(
            cur, bufA, P(0), P(1), P(2), P(3), P(4), P(5),
            B, cin, Tin, cout, Tout, s, d);

        const float* wds  = dsL[L] ? P(12) : nullptr;
        const float* gds  = dsL[L] ? P(13) : nullptr;
        const float* beds = dsL[L] ? P(14) : nullptr;
        const float* mds  = dsL[L] ? P(15) : nullptr;
        const float* vds  = dsL[L] ? P(16) : nullptr;
        conv5_bn_add_relu<<<(tot1 + 255) / 256, 256, 0, stream>>>(
            bufA, cur, nxt, P(6), P(7), P(8), P(9), P(10), P(11),
            wds, gds, beds, mds, vds,
            B, cout, Tout, d, cin, Tin, s, dsL[L]);

        cur = nxt;
        nxt = (nxt == buf0) ? buf1 : buf0;
        Tin = Tout;
    }

    // regressor: cur is [B,32,F]
    int totr = B * 6 * F;
    regressor_sigmoid<<<(totr + 255) / 256, 256, 0, stream>>>(
        cur, (const float*)d_in[115], (const float*)d_in[116], h6, B, 32, F, 6);

    // FIR + einsum (WMMA) + volume gate: one wave per (b,f)
    fir_wmma<<<(B * F) / 8, 256, 0, stream>>>(x, h6, (float*)d_out, B, T, F);
}
